// UnifiedModelRNN_32023276159204
// MI455X (gfx1250) — matrix operand order repaired and checked by reading, not compiled, not run on hardware
//
#include <hip/hip_runtime.h>

// ---------------------------------------------------------------------------
// CDNA5 (gfx1250) persistent LSTM kernel.
// Shapes: B=256, S=512, I=64, H=256, 4H=1024.
//  - pack_weights: one-shot f32 -> bf16 conversion of all weights into WMMA
//    B-panel layout (32x16 K-major tiles, per-lane contiguous 32B) in d_ws,
//    plus fused gate bias (b_ih + b_hh).
//  - rnn_main: 16 workgroups x 256 threads; each workgroup owns 16 batch rows
//    for all 512 steps (recurrence is per-row -> no inter-WG sync).
//    h / x_t / x_orig / ar_hidden live in LDS as bf16 A-panel tiles; the fp32
//    cell state c lives in registers (each wave owns the same 32 hidden
//    columns every step, with i/f/g/o of a cell in the same lane/VGPR slot).
//  - All GEMMs use v_wmma_f32_16x16x32_bf16 (f32 accumulate).
//  - Weight pointers are laundered through an SGPR asm barrier inside the
//    t-loop (blocks LICM/spill-to-scratch) AND cast to address space 1 so the
//    loads are GLOBAL class (LOADcnt only), not FLAT (LOADcnt+DScnt): this
//    decouples the L2 weight stream from the LDS pipeline waits.
// ---------------------------------------------------------------------------

typedef __bf16 bf16;
typedef bf16 v16bf __attribute__((ext_vector_type(16)));
typedef float v8f  __attribute__((ext_vector_type(8)));

// Global (address space 1) qualified element types.
typedef __attribute__((address_space(1))) unsigned short gus;
typedef __attribute__((address_space(1))) float gf32;
typedef __attribute__((address_space(1))) v16bf gv16bf;

constexpr int IN  = 64;    // input size
constexpr int NB  = 256;   // batch
constexpr int SEQ = 512;   // sequence length

// ---- pointer launder + AS(1) cast: block LICM, force global_load ----------
__device__ __forceinline__ const gus* opq_w(const unsigned short* p) {
  unsigned long long u = (unsigned long long)p;
  asm volatile("" : "+s"(u));
  return (const gus*)u;
}
__device__ __forceinline__ const gf32* opq_f(const float* p) {
  unsigned long long u = (unsigned long long)p;
  asm volatile("" : "+s"(u));
  return (const gf32*)u;
}
// Plain AS(1) reinterpret (no launder) via integer round-trip.
__device__ __forceinline__ const gf32* as_g(const float* p) {
  return (const gf32*)(unsigned long long)p;
}
__device__ __forceinline__ gf32* as_g(float* p) {
  return (gf32*)(unsigned long long)p;
}

// 32-byte WMMA operand load from a packed global panel.
__device__ __forceinline__ v16bf ldg_tile(const gus* p) {
  return *(const gv16bf*)p;
}

// ---- bf16 helpers ---------------------------------------------------------
__device__ __forceinline__ unsigned short f32_to_bf16_bits(float f) {
  union { float f; unsigned u; } v; v.f = f;
  unsigned r = v.u + 0x7FFFu + ((v.u >> 16) & 1u);   // round-to-nearest-even
  return (unsigned short)(r >> 16);
}

// B-panel pack: element (k, n) of a KxN operand.  WMMA bf16 B 32x16 layout:
// lane = (n%16) + 16*((k%16)>=8), per-lane slot j = (k%8) + 8*((k%32)>=16),
// tiles are [n/16][k/32] with nKt k-tiles per n-tile column.
__device__ __forceinline__ void pack_b(unsigned short* dst, float val,
                                       int n, int k, int nKt) {
  int lane = (n & 15) + (((k & 15) >= 8) ? 16 : 0);
  int j = (k & 7) + (((k & 31) >= 16) ? 8 : 0);
  int idx = (((n >> 4) * nKt + (k >> 5)) * 32 + lane) * 16 + j;
  dst[idx] = f32_to_bf16_bits(val);
}

// A-panel pack: element (m, k) of a 16xK operand.  WMMA bf16 A 16x32 layout:
// lane = m + 16*((k%16)>=8), per-lane slot j = (k%8) + 8*((k%32)>=16).
__device__ __forceinline__ void pack_a(unsigned short (*A)[32][16],
                                       int m, int k, float v) {
  int lane = m + (((k & 15) >= 8) ? 16 : 0);
  int j = (k & 7) + (((k & 31) >= 16) ? 8 : 0);
  A[k >> 5][lane][j] = f32_to_bf16_bits(v);
}

__device__ __forceinline__ float sigmoidf_(float x) {
  return 1.0f / (1.0f + __expf(-x));
}

// ---------------------------------------------------------------------------
// Weight packing kernel (runs once per launch; tiny).
// ---------------------------------------------------------------------------
__global__ __launch_bounds__(256) void pack_weights(
    const float* __restrict__ w_ih,   // (1024, 64)
    const float* __restrict__ w_hh,   // (1024, 256)
    const float* __restrict__ b_ih,   // (1024)
    const float* __restrict__ b_hh,   // (1024)
    const float* __restrict__ pit_w1, // (64, 256)
    const float* __restrict__ time_w1,// (64, 257)
    const float* __restrict__ ar_w1,  // (64, 66)
    const float* __restrict__ ar_w2,  // (64, 64)
    unsigned short* __restrict__ Wih, unsigned short* __restrict__ Whh,
    unsigned short* __restrict__ Pit1, unsigned short* __restrict__ Tim1,
    unsigned short* __restrict__ Ar1, unsigned short* __restrict__ Ar2,
    float* __restrict__ biasg) {
  int id = blockIdx.x * 256 + threadIdx.x;
  if (id < 65536) {                // W_ih^T : K=64, N=1024, nKt=2
    int n = id >> 6, k = id & 63;
    pack_b(Wih, w_ih[n * 64 + k], n, k, 2);
    return;
  }
  id -= 65536;
  if (id < 262144) {               // W_hh^T : K=256, N=1024, nKt=8
    int n = id >> 8, k = id & 255;
    pack_b(Whh, w_hh[n * 256 + k], n, k, 8);
    return;
  }
  id -= 262144;
  if (id < 16384) {                // pit_w1^T : K=256, N=64, nKt=8
    int n = id >> 8, k = id & 255;
    pack_b(Pit1, pit_w1[n * 256 + k], n, k, 8);
    return;
  }
  id -= 16384;
  if (id < 16384) {                // time_w1^T cols 0..255 : K=256, N=64
    int n = id >> 8, k = id & 255;
    pack_b(Tim1, time_w1[n * 257 + k], n, k, 8);
    return;
  }
  id -= 16384;
  if (id < 4096) {                 // ar_w1^T cols 2..65 : K=64, N=64, nKt=2
    int n = id >> 6, k = id & 63;
    pack_b(Ar1, ar_w1[n * 66 + 2 + k], n, k, 2);
    return;
  }
  id -= 4096;
  if (id < 4096) {                 // ar_w2^T : K=64, N=64, nKt=2
    int n = id >> 6, k = id & 63;
    pack_b(Ar2, ar_w2[n * 64 + k], n, k, 2);
    return;
  }
  id -= 4096;
  if (id < 1024) {                 // fused gate bias
    biasg[id] = b_ih[id] + b_hh[id];
  }
}

// ---------------------------------------------------------------------------
// Main persistent kernel: 16 blocks x 256 threads (8 wave32's).
// ---------------------------------------------------------------------------
__global__ __launch_bounds__(256) void rnn_main(
    const float* __restrict__ xG,             // (256, 512, 64)
    const unsigned short* __restrict__ WihG,
    const unsigned short* __restrict__ WhhG,
    const unsigned short* __restrict__ Pit1G,
    const unsigned short* __restrict__ Tim1G,
    const unsigned short* __restrict__ Ar1G,
    const unsigned short* __restrict__ Ar2G,
    const float* __restrict__ biasgG,         // (1024) fused b_ih+b_hh
    const float* __restrict__ pit_b1G,        // (64)
    const float* __restrict__ pit_w2G,        // (64)
    const float* __restrict__ pit_b2,         // (1)
    const float* __restrict__ time_w1G,       // (64,257) -> last column
    const float* __restrict__ time_b1G,       // (64)
    const float* __restrict__ time_w2G,       // (64)
    const float* __restrict__ time_b2,        // (1)
    const float* __restrict__ ar_w1G,         // (64,66) -> cols 0,1
    const float* __restrict__ ar_b1G,         // (64)
    const float* __restrict__ ar_b2G,         // (64)
    float* __restrict__ outG) {
  // LDS-resident per-step state (bf16 A-panel tiles + f32 staging).
  __shared__ alignas(32) unsigned short hA[8][32][16];   // h   (16 x 256)
  __shared__ alignas(32) unsigned short xA[2][32][16];   // x_t (16 x 64)
  __shared__ alignas(32) unsigned short xoA[2][32][16];  // x_orig_t
  __shared__ alignas(32) unsigned short arA[2][32][16];  // relu'd ar hidden
  __shared__ float ph[16 * 64];                          // pit hidden (relu)
  __shared__ float th[16 * 64];                          // time hidden (raw)
  __shared__ float pit_s[16];
  __shared__ float time_s[16];

  const int tid = threadIdx.x;
  const int w = tid >> 5;        // wave id 0..7
  const int lane = tid & 31;
  const int m0 = blockIdx.x * 16;

  const gf32* x = as_g(xG);
  gf32* out_pit  = as_g(outG);
  gf32* out_time = as_g(outG) + NB * SEQ;
  gf32* out_ar   = as_g(outG) + 2 * NB * SEQ;

  // h = 0
  for (int i = tid; i < 8 * 32 * 16; i += 256) ((unsigned short*)hA)[i] = 0;

  // Persistent fp32 cell state: wave w owns hidden columns w*32..w*32+31;
  // cs[tt][e] is cell (m = e + 8*(lane>=16), n = w*32 + tt*16 + lane%16).
  float cs[2][8];
  for (int a = 0; a < 2; ++a)
    for (int e = 0; e < 8; ++e) cs[a][e] = 0.0f;

  const v8f vzero = {};

  for (int t = 0; t < SEQ; ++t) {
    // Launder loop-invariant pointers so weight loads stay inside the loop
    // (streamed from L2) instead of being hoisted + spilled to scratch, and
    // tag them as global so loads are GLOBAL class (LOADcnt only, no DScnt).
    const gus* Wih  = opq_w(WihG);
    const gus* Whh  = opq_w(WhhG);
    const gus* Pit1 = opq_w(Pit1G);
    const gus* Tim1 = opq_w(Tim1G);
    const gus* Ar1  = opq_w(Ar1G);
    const gus* Ar2  = opq_w(Ar2G);
    const gf32* biasg   = opq_f(biasgG);
    const gf32* pit_b1  = opq_f(pit_b1G);
    const gf32* pit_w2  = opq_f(pit_w2G);
    const gf32* time_w1 = opq_f(time_w1G);
    const gf32* time_b1 = opq_f(time_b1G);
    const gf32* time_w2 = opq_f(time_w2G);
    const gf32* ar_w1   = opq_f(ar_w1G);
    const gf32* ar_b1   = opq_f(ar_b1G);
    const gf32* ar_b2   = opq_f(ar_b2G);

    // (A) stage x_orig_t into xoA (and seed xA / ar_out[:,0,:] at t==0).
    for (int q = 0; q < 4; ++q) {
      int e = tid * 4 + q;                   // 0..1023
      int m = e >> 6;
      int k = e & 63;
      float v = x[((m0 + m) * SEQ + t) * IN + k];
      pack_a(xoA, m, k, v);
      if (t == 0) {
        pack_a(xA, m, k, v);
        out_ar[((m0 + m) * SEQ + 0) * IN + k] = v;
      }
    }
    __syncthreads();

    // (B) gates = x_t@W_ih^T + h@W_hh^T   (each wave: 8 gate tiles).
    v8f acc[4][2];
    for (int g4 = 0; g4 < 4; ++g4)
      for (int tt = 0; tt < 2; ++tt) acc[g4][tt] = vzero;

    for (int tt = 0; tt < 2; ++tt) {
      for (int g4 = 0; g4 < 4; ++g4) {
        int nt = g4 * 16 + w * 2 + tt;        // n-tile in 0..63
        v8f c = acc[g4][tt];
        for (int kt = 0; kt < 2; ++kt) {      // x part, K = 64
          v16bf a = *(const v16bf*)&xA[kt][lane][0];
          v16bf b = ldg_tile(Wih + ((nt * 2 + kt) * 32 + lane) * 16);
          c = __builtin_amdgcn_wmma_f32_16x16x32_bf16(false, a, false, b,
                                                      (short)0, c, false, false);
        }
        for (int kt = 0; kt < 8; ++kt) {      // h part, K = 256
          v16bf a = *(const v16bf*)&hA[kt][lane][0];
          v16bf b = ldg_tile(Whh + ((nt * 8 + kt) * 32 + lane) * 16);
          c = __builtin_amdgcn_wmma_f32_16x16x32_bf16(false, a, false, b,
                                                      (short)0, c, false, false);
        }
        acc[g4][tt] = c;
      }
    }
    __syncthreads();   // all hA/xA reads complete before overwriting hA

    // (C) LSTM pointwise entirely in registers; write new h to hA (bf16).
    for (int tt = 0; tt < 2; ++tt) {
      int nb = w * 32 + tt * 16 + (lane & 15);   // hidden unit 0..255
      for (int e = 0; e < 8; ++e) {
        int m = e + ((lane >> 4) << 3);
        float ig = sigmoidf_(acc[0][tt][e] + biasg[nb]);
        float fg = sigmoidf_(acc[1][tt][e] + biasg[256 + nb]);
        float gg = tanhf(acc[2][tt][e] + biasg[512 + nb]);
        float og = sigmoidf_(acc[3][tt][e] + biasg[768 + nb]);
        float cn = fg * cs[tt][e] + ig * gg;
        cs[tt][e] = cn;
        float hh = og * tanhf(cn);
        pack_a(hA, m, nb, hh);
      }
    }
    __syncthreads();

    // (D) pit hidden (waves 0-3) / time hidden partial (waves 4-7), K=256.
    {
      int nt = w & 3;
      const gus* Wp = (w < 4) ? Pit1 : Tim1;
      v8f c = vzero;
      for (int kt = 0; kt < 8; ++kt) {
        v16bf a = *(const v16bf*)&hA[kt][lane][0];
        v16bf b = ldg_tile(Wp + ((nt * 8 + kt) * 32 + lane) * 16);
        c = __builtin_amdgcn_wmma_f32_16x16x32_bf16(false, a, false, b,
                                                    (short)0, c, false, false);
      }
      const gf32* b1 = (w < 4) ? pit_b1 : time_b1;
      float* dst = (w < 4) ? ph : th;
      for (int e = 0; e < 8; ++e) {
        int m = e + ((lane >> 4) << 3);
        int n = nt * 16 + (lane & 15);
        float v = c[e] + b1[n];
        if (w < 4) v = fmaxf(v, 0.0f);   // pit: relu now; time: relu later
        dst[m * 64 + n] = v;
      }
    }
    __syncthreads();

    // (E) pit & time scalar heads (64-length dots), 16 lanes of wave 0.
    if (w == 0 && lane < 16) {
      int m = lane;
      float p = pit_b2[0];
      for (int j = 0; j < 64; ++j) p += ph[m * 64 + j] * pit_w2[j];
      pit_s[m] = p;
      out_pit[(m0 + m) * SEQ + t] = p;

      float tm = time_b2[0];
      for (int j = 0; j < 64; ++j) {
        float v = th[m * 64 + j] + p * time_w1[j * 257 + 256];
        v = fmaxf(v, 0.0f);
        tm += v * time_w2[j];
      }
      time_s[m] = tm;
      out_time[(m0 + m) * SEQ + t] = tm;
    }
    __syncthreads();

    // (F) ar hidden: x_orig part via WMMA + rank-1 pit/time terms + relu.
    if (w < 4) {
      int nt = w;
      v8f c = vzero;
      for (int kt = 0; kt < 2; ++kt) {
        v16bf a = *(const v16bf*)&xoA[kt][lane][0];
        v16bf b = ldg_tile(Ar1 + ((nt * 2 + kt) * 32 + lane) * 16);
        c = __builtin_amdgcn_wmma_f32_16x16x32_bf16(false, a, false, b,
                                                    (short)0, c, false, false);
      }
      for (int e = 0; e < 8; ++e) {
        int m = e + ((lane >> 4) << 3);
        int n = nt * 16 + (lane & 15);
        float v = c[e] + ar_b1[n] + pit_s[m] * ar_w1[n * 66 + 0]
                                  + time_s[m] * ar_w1[n * 66 + 1];
        v = fmaxf(v, 0.0f);
        pack_a(arA, m, n, v);
      }
    }
    __syncthreads();

    // (G) nxt = relu_hidden @ ar_w2^T; emit ar_out[:, t+1, :], feed xA.
    if (w < 4) {
      int nt = w;
      v8f c = vzero;
      for (int kt = 0; kt < 2; ++kt) {
        v16bf a = *(const v16bf*)&arA[kt][lane][0];
        v16bf b = ldg_tile(Ar2 + ((nt * 2 + kt) * 32 + lane) * 16);
        c = __builtin_amdgcn_wmma_f32_16x16x32_bf16(false, a, false, b,
                                                    (short)0, c, false, false);
      }
      for (int e = 0; e < 8; ++e) {
        int m = e + ((lane >> 4) << 3);
        int n = nt * 16 + (lane & 15);
        float v = c[e] + ar_b2[n];
        if (t < SEQ - 1)
          out_ar[((m0 + m) * SEQ + (t + 1)) * IN + n] = v;
        pack_a(xA, m, n, v);
      }
    }
    __syncthreads();
  }
}

// ---------------------------------------------------------------------------
extern "C" void kernel_launch(void* const* d_in, const int* in_sizes, int n_in,
                              void* d_out, int out_size, void* d_ws,
                              size_t ws_size, hipStream_t stream) {
  const float* x       = (const float*)d_in[0];
  const float* w_ih    = (const float*)d_in[1];
  const float* w_hh    = (const float*)d_in[2];
  const float* b_ih    = (const float*)d_in[3];
  const float* b_hh    = (const float*)d_in[4];
  const float* pit_w1  = (const float*)d_in[5];
  const float* pit_b1  = (const float*)d_in[6];
  const float* pit_w2  = (const float*)d_in[7];
  const float* pit_b2  = (const float*)d_in[8];
  const float* time_w1 = (const float*)d_in[9];
  const float* time_b1 = (const float*)d_in[10];
  const float* time_w2 = (const float*)d_in[11];
  const float* time_b2 = (const float*)d_in[12];
  const float* ar_w1   = (const float*)d_in[13];
  const float* ar_b1   = (const float*)d_in[14];
  const float* ar_w2   = (const float*)d_in[15];
  const float* ar_b2   = (const float*)d_in[16];

  // Workspace layout (bf16-packed weight panels + fused bias), ~724 KB.
  unsigned short* Wih  = (unsigned short*)d_ws;   // 65536 bf16
  unsigned short* Whh  = Wih + 65536;             // 262144 bf16
  unsigned short* Pit1 = Whh + 262144;            // 16384 bf16
  unsigned short* Tim1 = Pit1 + 16384;            // 16384 bf16
  unsigned short* Ar1  = Tim1 + 16384;            // 4096 bf16
  unsigned short* Ar2  = Ar1 + 4096;              // 4096 bf16
  float* biasg         = (float*)(Ar2 + 4096);    // 1024 f32

  // 65536+262144+16384+16384+4096+4096+1024 = 369664 work items.
  pack_weights<<<dim3(369664 / 256), dim3(256), 0, stream>>>(
      w_ih, w_hh, b_ih, b_hh, pit_w1, time_w1, ar_w1, ar_w2,
      Wih, Whh, Pit1, Tim1, Ar1, Ar2, biasg);

  rnn_main<<<dim3(NB / 16), dim3(256), 0, stream>>>(
      x, Wih, Whh, Pit1, Tim1, Ar1, Ar2, biasg,
      pit_b1, pit_w2, pit_b2, time_w1, time_b1, time_w2, time_b2,
      ar_w1, ar_b1, ar_b2, (float*)d_out);

  (void)in_sizes; (void)n_in; (void)out_size; (void)ws_size;
}
